// MSDeformableAttention3D_64707977282319
// MI455X (gfx1250) — matrix operand order repaired and checked by reading, not compiled, not run on hardware
//
#include <hip/hip_runtime.h>

typedef _Float16 h8   __attribute__((ext_vector_type(8)));
typedef _Float16 v16h __attribute__((ext_vector_type(16)));
typedef float    v8f  __attribute__((ext_vector_type(8)));

#define BM 128
#define BN 128
#define BK 32

// ---------------------------------------------------------------------------
// Elementwise f32 -> f16 (8 elems/thread, b128 in / b128 out)
// ---------------------------------------------------------------------------
__global__ __launch_bounds__(256)
void cvt_f16_k(const float* __restrict__ in, _Float16* __restrict__ out, int n8)
{
    const int i = blockIdx.x * blockDim.x + threadIdx.x;
    if (i >= n8) return;
    const float4* p = (const float4*)in + (size_t)i * 2;
    const float4 a = p[0], b = p[1];
    h8 o;
    o[0] = (_Float16)a.x; o[1] = (_Float16)a.y; o[2] = (_Float16)a.z; o[3] = (_Float16)a.w;
    o[4] = (_Float16)b.x; o[5] = (_Float16)b.y; o[6] = (_Float16)b.z; o[7] = (_Float16)b.w;
    *((h8*)out + i) = o;
}

// ---------------------------------------------------------------------------
// Tiled transpose + convert: in f32 [R][C] -> out f16 [C][R]
// (weights [K][N] -> [N][K]; channel-major feats [256][HW] -> [HW][256])
// ---------------------------------------------------------------------------
__global__ __launch_bounds__(256)
void transpose_cvt_k(const float* __restrict__ in, _Float16* __restrict__ out,
                     int R, int C)
{
    __shared__ float t[32][33];
    const int bx = blockIdx.x * 32;   // col base
    const int by = blockIdx.y * 32;   // row base
    const int tx = threadIdx.x, ty = threadIdx.y;   // 32 x 8
#pragma unroll
    for (int j = 0; j < 32; j += 8) {
        const int r = by + ty + j, c = bx + tx;
        t[ty + j][tx] = (r < R && c < C) ? in[(size_t)r * C + c] : 0.0f;
    }
    __syncthreads();
#pragma unroll
    for (int j = 0; j < 32; j += 8) {
        const int c = bx + ty + j, r = by + tx;
        if (c < C && r < R) out[(size_t)c * R + r] = (_Float16)t[tx][ty + j];
    }
}

// ---------------------------------------------------------------------------
// LDS-free direct-fragment WMMA GEMM:
//   D[M][N] = A[M][K] (f16 row-major) x Bt[N][K]^T (f16 n-major) + bias
// Fragments loaded straight from global memory (L2-resident operands):
//   A frag: 2x global_load_b128 per 16x32 tile; B frag: 2x global_load_b128.
//   MODE 0: Out[row*N + col]
//   MODE 1: Out[((col>>5)*M + row)*32 + (col&31)]  (value tensor [head][s][hd])
// ---------------------------------------------------------------------------
template<int MODE>
__global__ __launch_bounds__(256)
void gemm_wmma_direct(const _Float16* __restrict__ A,
                      const _Float16* __restrict__ Bt,
                      const float* __restrict__ bias,
                      float* __restrict__ Out, int M, int N, int K)
{
    const int tid  = threadIdx.x;
    const int lane = tid & 31;
    const int wave = tid >> 5;
    const int ntn  = N / BN;
    const int bx   = blockIdx.x % ntn;
    const int by   = blockIdx.x / ntn;
    const int row0 = by * BM;
    const int col0 = bx * BN;
    const int wm   = wave & 3;    // 4 waves along M (4*32 = 128)
    const int wn   = wave >> 2;   // 2 waves along N (2*64 = 128)
    const int lr   = lane & 15;
    const int kg   = lane >> 4;

    v8f acc[2][4];
#pragma unroll
    for (int i = 0; i < 2; ++i)
#pragma unroll
        for (int j = 0; j < 4; ++j) acc[i][j] = (v8f)0.0f;

    // base pointers per fragment (OOB rows clamp-loaded, masked at store)
    const _Float16* aBase[2];
#pragma unroll
    for (int mt = 0; mt < 2; ++mt) {
        int r = row0 + wm * 32 + mt * 16 + lr;
        r = (r < M) ? r : (M - 1);
        aBase[mt] = A + (size_t)r * K;
    }
    const _Float16* bBase[4];
#pragma unroll
    for (int nt = 0; nt < 4; ++nt) {
        const int n = col0 + wn * 64 + nt * 16 + lr;
        bBase[nt] = Bt + (size_t)n * K;
    }

#pragma unroll 2
    for (int k0 = 0; k0 < K; k0 += BK) {
        v16h af[2], bf[4];
#pragma unroll
        for (int mt = 0; mt < 2; ++mt) {
            union { v16h v; h8 h[2]; } u;
            u.h[0] = *(const h8*)(aBase[mt] + k0 + kg * 8);        // K = kg*8 + 0..7
            u.h[1] = *(const h8*)(aBase[mt] + k0 + 16 + kg * 8);   // K = 16 + kg*8 + 0..7
            af[mt] = u.v;
        }
#pragma unroll
        for (int nt = 0; nt < 4; ++nt) {
            union { v16h v; h8 h[2]; } u;
            u.h[0] = *(const h8*)(bBase[nt] + k0 + kg * 8);        // K = kg*8 + 0..7
            u.h[1] = *(const h8*)(bBase[nt] + k0 + 16 + kg * 8);   // K = 16 + kg*8 + 0..7
            bf[nt] = u.v;
        }
#pragma unroll
        for (int mt = 0; mt < 2; ++mt)
#pragma unroll
            for (int nt = 0; nt < 4; ++nt)
                acc[mt][nt] = __builtin_amdgcn_wmma_f32_16x16x32_f16(
                    false, af[mt], false, bf[nt],
                    (short)0, acc[mt][nt], false, false);
    }

    // epilogue: C/D layout -> M = i + 8*kg, N = lr
#pragma unroll
    for (int mt = 0; mt < 2; ++mt) {
#pragma unroll
        for (int nt = 0; nt < 4; ++nt) {
            const int colg = col0 + wn * 64 + nt * 16 + lr;
            const float bv = bias[colg];
#pragma unroll
            for (int i = 0; i < 8; ++i) {
                const int rowg = row0 + wm * 32 + mt * 16 + kg * 8 + i;
                if (rowg < M) {
                    const float val = acc[mt][nt][i] + bv;
                    if (MODE == 0) {
                        Out[(size_t)rowg * N + colg] = val;
                    } else {
                        const int hh = colg >> 5, d = colg & 31;
                        Out[((size_t)hh * M + rowg) * 32 + d] = val;
                    }
                }
            }
        }
    }
}

// ---------------------------------------------------------------------------
// Deformable sampling: one wave per (query, head); lane = hd channel (hd==32).
// Corner gathers are coalesced 128B transactions; value tensors are L2-resident.
// Writes agg as f16 (feeds the final WMMA GEMM directly).
// ---------------------------------------------------------------------------
__global__ __launch_bounds__(256)
void deform_sample_k(const float* __restrict__ off,   // [Nq,8,4,8,2]
                     const float* __restrict__ attn,  // [Nq,8,4,8] raw logits
                     const float* __restrict__ refp,  // [Nq,4,2]
                     const float* __restrict__ vbuf,  // per-level [hh][HW][32], concat
                     _Float16* __restrict__ agg,      // [Nq,256] f16
                     int Nq)
{
    const int gwave = blockIdx.x * 8 + (threadIdx.x >> 5);
    const int lane  = threadIdx.x & 31;
    const int q  = gwave >> 3;
    const int hh = gwave & 7;
    if (q >= Nq) return;

    const int dims[4]  = {128, 64, 32, 16};
    const int vbase[4] = {0, 16384 * 256, (16384 + 4096) * 256,
                          (16384 + 4096 + 1024) * 256};

    float acc = 0.0f;
#pragma unroll
    for (int l = 0; l < 4; ++l) {
        const int Wl = dims[l], Hl = dims[l];
        const int HW = Hl * Wl;
        const float* vl = vbuf + vbase[l] + hh * HW * 32;

        // softmax over P=8 (wave-uniform scalars)
        const float* ap = attn + ((size_t)q * 8 + hh) * 32 + l * 8;
        float wv[8], mx = -1e30f;
#pragma unroll
        for (int p = 0; p < 8; ++p) { wv[p] = ap[p]; mx = fmaxf(mx, wv[p]); }
        float ssum = 0.0f;
#pragma unroll
        for (int p = 0; p < 8; ++p) { wv[p] = __expf(wv[p] - mx); ssum += wv[p]; }
        const float inv = 1.0f / ssum;

        const float rx = refp[((size_t)q * 4 + l) * 2 + 0];
        const float ry = refp[((size_t)q * 4 + l) * 2 + 1];
        const float* op = off + (((size_t)q * 8 + hh) * 4 + l) * 16;

#pragma unroll
        for (int p = 0; p < 8; ++p) {
            const float sx = rx + tanhf(op[p * 2 + 0]) * 0.25f;
            const float sy = ry + tanhf(op[p * 2 + 1]) * 0.25f;
            const float x  = (sx + 1.0f) * 0.5f * (float)(Wl - 1);
            const float y  = (sy + 1.0f) * 0.5f * (float)(Hl - 1);
            const float x0 = floorf(x), y0 = floorf(y);
            const float fx = x - x0,   fy = y - y0;
            const int ix0 = (int)x0, iy0 = (int)y0, ix1 = ix0 + 1, iy1 = iy0 + 1;
            const bool bx0 = (ix0 >= 0) & (ix0 < Wl), bx1 = (ix1 >= 0) & (ix1 < Wl);
            const bool by0 = (iy0 >= 0) & (iy0 < Hl), by1 = (iy1 >= 0) & (iy1 < Hl);
            float v00 = 0.f, v01 = 0.f, v10 = 0.f, v11 = 0.f;
            if (bx0 & by0) v00 = vl[(iy0 * Wl + ix0) * 32 + lane];
            if (bx1 & by0) v01 = vl[(iy0 * Wl + ix1) * 32 + lane];
            if (bx0 & by1) v10 = vl[(iy1 * Wl + ix0) * 32 + lane];
            if (bx1 & by1) v11 = vl[(iy1 * Wl + ix1) * 32 + lane];
            const float bil = v00 * (1.0f - fx) * (1.0f - fy)
                            + v01 * fx          * (1.0f - fy)
                            + v10 * (1.0f - fx) * fy
                            + v11 * fx          * fy;
            acc += (wv[p] * inv) * bil;
        }
    }
    agg[(size_t)q * 256 + hh * 32 + lane] = (_Float16)acc;
}

// ---------------------------------------------------------------------------
extern "C" void kernel_launch(void* const* d_in, const int* in_sizes, int n_in,
                              void* d_out, int out_size, void* d_ws, size_t ws_size,
                              hipStream_t stream)
{
    (void)in_sizes; (void)n_in; (void)out_size; (void)ws_size;

    const float* query  = (const float*)d_in[0];
    const float* refp   = (const float*)d_in[1];
    const float* feat[4] = {(const float*)d_in[2], (const float*)d_in[3],
                            (const float*)d_in[4], (const float*)d_in[5]};
    const float* W_off  = (const float*)d_in[6];
    const float* b_off  = (const float*)d_in[7];
    const float* W_attn = (const float*)d_in[8];
    const float* b_attn = (const float*)d_in[9];
    const float* W_val  = (const float*)d_in[10];
    const float* b_val  = (const float*)d_in[11];
    const float* W_out  = (const float*)d_in[12];
    const float* b_out  = (const float*)d_in[13];

    const int Nq = 40000, C = 256;
    const int HWs[4] = {16384, 4096, 1024, 256};
    const int HWtot  = 16384 + 4096 + 1024 + 256;   // 21760

    // workspace layout (bytes, 256B aligned)
    char* ws = (char*)d_ws;
    float*     off   = (float*)    (ws + 0L);            // Nq*512*4   = 81,920,000
    float*     attn  = (float*)    (ws + 81920000L);     // Nq*256*4   = 40,960,000
    float*     vbuf  = (float*)    (ws + 122880000L);    // 21760*256*4= 22,282,240
    _Float16*  aggh  = (_Float16*) (ws + 145162240L);    // Nq*256*2   = 20,480,000
    _Float16*  Qh    = (_Float16*) (ws + 165642240L);    // Nq*256*2   = 20,480,000
    _Float16*  Fh    = (_Float16*) (ws + 186122240L);    // 21760*256*2= 11,141,120
    _Float16*  WhOff = (_Float16*) (ws + 197263360L);    // 512*256*2  =    262,144
    _Float16*  WhAtt = (_Float16*) (ws + 197525504L);    // 256*256*2  =    131,072
    _Float16*  WhVal = (_Float16*) (ws + 197656576L);    // 256*256*2  =    131,072
    _Float16*  WhOut = (_Float16*) (ws + 197787648L);    // 256*256*2  =    131,072

    // ---- one-time repack: precision + layout ----
    {   // query f32 -> f16 (same layout)
        const int n8 = Nq * C / 8;
        cvt_f16_k<<<(n8 + 255) / 256, 256, 0, stream>>>(query, Qh, n8);
    }
    // weights [K][N] -> f16 [N][K]
    transpose_cvt_k<<<dim3(512 / 32, C / 32), dim3(32, 8), 0, stream>>>(W_off,  WhOff, C, 512);
    transpose_cvt_k<<<dim3(C   / 32, C / 32), dim3(32, 8), 0, stream>>>(W_attn, WhAtt, C, C);
    transpose_cvt_k<<<dim3(C   / 32, C / 32), dim3(32, 8), 0, stream>>>(W_val,  WhVal, C, C);
    transpose_cvt_k<<<dim3(C   / 32, C / 32), dim3(32, 8), 0, stream>>>(W_out,  WhOut, C, C);
    // feats [256][HW] -> f16 [HW][256]
    {
        long fofs = 0;
        for (int l = 0; l < 4; ++l) {
            transpose_cvt_k<<<dim3(HWs[l] / 32, C / 32), dim3(32, 8), 0, stream>>>(
                feat[l], Fh + fofs, C, HWs[l]);
            fofs += (long)HWs[l] * C;
        }
    }

    auto grid_of = [](int M, int N) { return ((M + BM - 1) / BM) * (N / BN); };

    // offsets: Qh @ W_off^T + b_off   [Nq,512]
    gemm_wmma_direct<0><<<grid_of(Nq, 512), 256, 0, stream>>>(
        Qh, WhOff, b_off, off, Nq, 512, C);
    // attention logits: Qh @ W_attn^T + b_attn   [Nq,256]
    gemm_wmma_direct<0><<<grid_of(Nq, 256), 256, 0, stream>>>(
        Qh, WhAtt, b_attn, attn, Nq, 256, C);
    // value projection per level -> [head][s][hd] (f32)
    {
        long fofs = 0, vofs = 0;
        for (int l = 0; l < 4; ++l) {
            gemm_wmma_direct<1><<<grid_of(HWs[l], 256), 256, 0, stream>>>(
                Fh + fofs, WhVal, b_val, vbuf + vofs, HWs[l], 256, C);
            fofs += (long)HWs[l] * C;
            vofs += (long)HWs[l] * C;
        }
        (void)HWtot;
    }
    // bilinear sampling + softmax-weighted aggregation -> agg f16 [Nq,256]
    deform_sample_k<<<Nq, 256, 0, stream>>>(off, attn, refp, vbuf, aggh, Nq);
    // output projection: agg @ W_out^T + b_out -> d_out
    gemm_wmma_direct<0><<<grid_of(Nq, 256), 256, 0, stream>>>(
        aggh, WhOut, b_out, (float*)d_out, Nq, 256, C);
}
